// MSFNet_29884382446013
// MI455X (gfx1250) — matrix operand order repaired and checked by reading, not compiled, not run on hardware
//
#include <hip/hip_runtime.h>

// Problem constants
#define BB   256
#define NN   256            // S+1 tokens
#define SS   255
#define DD   512
#define DQK  64
#define H1   1024
#define H2   512
#define OUTC 128

typedef __attribute__((ext_vector_type(16))) _Float16 v16h;
typedef __attribute__((ext_vector_type(8)))  _Float16 v8h;
typedef __attribute__((ext_vector_type(8)))  float    v8f;
typedef __attribute__((ext_vector_type(4)))  float    v4f;

__device__ __forceinline__ v16h mk16(v8h lo, v8h hi) {
    return __builtin_shufflevector(lo, hi, 0,1,2,3,4,5,6,7,8,9,10,11,12,13,14,15);
}

// ---------------- f32 -> f16 convert ----------------
__global__ void f32_to_f16_kernel(const float* __restrict__ src, _Float16* __restrict__ dst, int n) {
    int i = blockIdx.x * blockDim.x + threadIdx.x;
    if (i < n) dst[i] = (_Float16)src[i];
}

// ---------------- copy f32 ----------------
__global__ void copy_f32_kernel(const float* __restrict__ src, float* __restrict__ dst, int n) {
    int i = blockIdx.x * blockDim.x + threadIdx.x;
    if (i < n) dst[i] = src[i];
}

// ---------------- generic tiled f32 GEMM: C[M,N] = A[M,K] @ B[K,N] ----------------
// Row-major; M,N multiples of 64, K multiple of 16. Tiny share of total work,
// exists to read shared weights O(1) times instead of once per batch.
__global__ __launch_bounds__(256) void gemm_f32_kernel(
    const float* __restrict__ A, const float* __restrict__ B, float* __restrict__ C,
    int M, int N, int K)
{
    __shared__ float As[64][17];
    __shared__ float Bs[16][65];
    const int nbx = N >> 6;
    const int bx = blockIdx.x % nbx, by = blockIdx.x / nbx;
    const int tx = threadIdx.x & 15, ty = threadIdx.x >> 4;

    float acc[4][4];
    for (int i = 0; i < 4; ++i) for (int j = 0; j < 4; ++j) acc[i][j] = 0.f;

    for (int k0 = 0; k0 < K; k0 += 16) {
        for (int q = 0; q < 4; ++q) {
            int idx = threadIdx.x + 256 * q;
            int r = idx >> 4, c = idx & 15;
            As[r][c] = A[(by * 64 + r) * K + k0 + c];
        }
        for (int q = 0; q < 4; ++q) {
            int idx = threadIdx.x + 256 * q;
            int r = idx >> 6, c = idx & 63;
            Bs[r][c] = B[(k0 + r) * N + bx * 64 + c];
        }
        __syncthreads();
        #pragma unroll
        for (int kk = 0; kk < 16; ++kk) {
            float ar[4], br[4];
            #pragma unroll
            for (int i = 0; i < 4; ++i) ar[i] = As[ty * 4 + i][kk];
            #pragma unroll
            for (int j = 0; j < 4; ++j) br[j] = Bs[kk][tx * 4 + j];
            #pragma unroll
            for (int i = 0; i < 4; ++i)
                #pragma unroll
                for (int j = 0; j < 4; ++j) acc[i][j] += ar[i] * br[j];
        }
        __syncthreads();
    }
    for (int i = 0; i < 4; ++i)
        for (int j = 0; j < 4; ++j)
            C[(by * 64 + ty * 4 + i) * N + bx * 64 + tx * 4 + j] = acc[i][j];
}

// ---------------- z[j] = gamma*(bv @ W1)[j] + b1[j] ----------------
__global__ void z_kernel(const float* __restrict__ bv, const float* __restrict__ W1,
                         const float* __restrict__ b1, const float* __restrict__ gammap,
                         float* __restrict__ z) {
    int j = blockIdx.x * 256 + threadIdx.x;
    float s = 0.f;
    for (int d = 0; d < DD; ++d) s += bv[d] * W1[d * H1 + j];
    z[j] = gammap[0] * s + b1[j];
}

// ---------------- per-batch attention scalars ----------------
// Logits reduce to a_m * (a_n*g + r); m-constant terms cancel in softmax.
// c_n = sum_m A[n,m] a_m ;  alpha_n = gamma * c_n.
__global__ __launch_bounds__(256) void attn_scalar_kernel(
    const float* __restrict__ Xs, const float* __restrict__ q0all,
    const float* __restrict__ k0all, const float* __restrict__ bq,
    const float* __restrict__ gammap,
    float* __restrict__ rowA, float* __restrict__ rowAl)
{
    __shared__ float aS[NN], q0s[DQK], k0s[DQK];
    const int b = blockIdx.x, t = threadIdx.x;
    if (t == 0) aS[0] = 1.f;
    if (t < SS) aS[t + 1] = Xs[b * SS + t];
    if (t < DQK) { q0s[t] = q0all[b * DQK + t]; k0s[t] = k0all[b * DQK + t]; }
    __syncthreads();

    float g = 0.f, r = 0.f;
    for (int i = 0; i < DQK; ++i) { g += q0s[i] * k0s[i]; r += bq[i] * k0s[i]; }

    const float an = aS[t];
    const float beta = an * g + r;
    float mx = -3.4e38f;
    for (int m = 0; m < NN; ++m) mx = fmaxf(mx, aS[m] * beta);
    float sum = 0.f, cs = 0.f;
    for (int m = 0; m < NN; ++m) {
        float am = aS[m];
        float e = __expf(am * beta - mx);
        sum += e; cs += am * e;
    }
    rowA [b * NN + t] = an;
    rowAl[b * NN + t] = gammap[0] * (cs / sum);
}

// ---------------- fused WMMA kernel: relu(relu(h1@W2+b2)@W3+b3) ----------------
// h1[r,k] = relu(alpha_r*w[k] + a_r*u[k] + z[k]) is built per-lane IN REGISTERS
// from broadcast v4f LDS reads -> barrier-free WMMA main loop.
// Block = 32 rows of one batch; 8 waves = 2 M-tiles x 4 N-chunks (128 cols each).
__global__ __launch_bounds__(256) void fused_mlp_kernel(
    const float* __restrict__ UW, const float* __restrict__ z,
    const float* __restrict__ rowA, const float* __restrict__ rowAl,
    const _Float16* __restrict__ W2h, const float* __restrict__ b2,
    const _Float16* __restrict__ W3h, const float* __restrict__ b3,
    float* __restrict__ out)
{
    __shared__ float __attribute__((aligned(16))) us[H1];
    __shared__ float __attribute__((aligned(16))) wsv[H1];
    __shared__ float __attribute__((aligned(16))) zs[H1];
    __shared__ _Float16 __attribute__((aligned(16))) h2s[32 * H2];

    const int t  = threadIdx.x;
    const int r0 = blockIdx.x * 32;         // global row base (row = b*256 + n)
    const int b  = r0 >> 8;

    const float* ub = UW + b * H1;           // u = x0@W1  (row b of UW)
    const float* wb = UW + (BB + b) * H1;    // w = v0@W1  (row 256+b of UW)
    for (int qq = 0; qq < 4; ++qq) {
        int j = t + 256 * qq;
        us[j] = ub[j]; wsv[j] = wb[j]; zs[j] = z[j];
    }
    __syncthreads();

    const int wv    = t >> 5;
    const int lane  = t & 31;
    const int lrow  = lane & 15;
    const int lhalf = lane >> 4;
    const int kb    = lhalf * 8;
    const int mt    = wv >> 2;              // 0..1 (16 rows each)
    const int nc    = wv & 3;               // 0..3 (128 cols each)

    // per-lane row scalars for A-fragment build
    const float alr = rowAl[r0 + mt * 16 + lrow];
    const float ar  = rowA [r0 + mt * 16 + lrow];

    // -------- GEMM1: [32 x 1024] @ [1024 x 512], K-steps of 32, no barriers --------
    v8f acc[8];
    for (int i = 0; i < 8; ++i) for (int j = 0; j < 8; ++j) acc[i][j] = 0.f;

    for (int ks = 0; ks < H1 / 32; ++ks) {
        const int k0 = ks * 32;
        const int ko = k0 + kb;             // lane's first K-octet base (mult of 8)

        v4f w0 = *(const v4f*)&wsv[ko],      w1 = *(const v4f*)&wsv[ko + 4];
        v4f w2 = *(const v4f*)&wsv[ko + 16], w3 = *(const v4f*)&wsv[ko + 20];
        v4f u0 = *(const v4f*)&us[ko],       u1 = *(const v4f*)&us[ko + 4];
        v4f u2 = *(const v4f*)&us[ko + 16],  u3 = *(const v4f*)&us[ko + 20];
        v4f z0 = *(const v4f*)&zs[ko],       z1 = *(const v4f*)&zs[ko + 4];
        v4f z2 = *(const v4f*)&zs[ko + 16],  z3 = *(const v4f*)&zs[ko + 20];

        v16h afrag;
        #pragma unroll
        for (int j = 0; j < 4; ++j) {
            afrag[j]      = (_Float16)fmaxf(alr * w0[j] + ar * u0[j] + z0[j], 0.f);
            afrag[4 + j]  = (_Float16)fmaxf(alr * w1[j] + ar * u1[j] + z1[j], 0.f);
            afrag[8 + j]  = (_Float16)fmaxf(alr * w2[j] + ar * u2[j] + z2[j], 0.f);
            afrag[12 + j] = (_Float16)fmaxf(alr * w3[j] + ar * u3[j] + z3[j], 0.f);
        }

        const _Float16* brow = W2h + (k0 + kb) * H2 + nc * 128 + lrow;
        #pragma unroll
        for (int tt = 0; tt < 8; ++tt) {
            const _Float16* bp = brow + tt * 16;
            v16h bfrag;
            #pragma unroll
            for (int s = 0; s < 8; ++s) { bfrag[s] = bp[s * H2]; bfrag[8 + s] = bp[(16 + s) * H2]; }
            acc[tt] = __builtin_amdgcn_wmma_f32_16x16x32_f16(
                false, afrag, false, bfrag, (short)0, acc[tt], false, false);
        }
    }

    // epilogue 1: +b2, relu, f16 -> LDS h2s[32][512]
    {
        const int mbase = mt * 16 + lhalf * 8;
        for (int tt = 0; tt < 8; ++tt) {
            int ncol = nc * 128 + tt * 16 + lrow;
            float bias = b2[ncol];
            #pragma unroll
            for (int i = 0; i < 8; ++i) {
                float vl = acc[tt][i] + bias;
                h2s[(mbase + i) * H2 + ncol] = (_Float16)fmaxf(vl, 0.f);
            }
        }
    }
    __syncthreads();

    // -------- GEMM2: [32 x 512] @ [512 x 128], K-steps of 32 --------
    v8f acc2[2];
    for (int i = 0; i < 2; ++i) for (int j = 0; j < 8; ++j) acc2[i][j] = 0.f;

    for (int ks = 0; ks < H2 / 32; ++ks) {
        const int k0 = ks * 32;
        const _Float16* ap = &h2s[(mt * 16 + lrow) * H2 + k0 + kb];
        v16h afrag = mk16(*(const v8h*)ap, *(const v8h*)(ap + 16));

        const _Float16* brow = W3h + (k0 + kb) * OUTC + nc * 32 + lrow;
        #pragma unroll
        for (int tt = 0; tt < 2; ++tt) {
            const _Float16* bp = brow + tt * 16;
            v16h bfrag;
            #pragma unroll
            for (int s = 0; s < 8; ++s) { bfrag[s] = bp[s * OUTC]; bfrag[8 + s] = bp[(16 + s) * OUTC]; }
            acc2[tt] = __builtin_amdgcn_wmma_f32_16x16x32_f16(
                false, afrag, false, bfrag, (short)0, acc2[tt], false, false);
        }
    }

    // epilogue 2: +b3, relu, f32 -> out[row*128 + col]
    {
        const int mbase = mt * 16 + lhalf * 8;
        for (int tt = 0; tt < 2; ++tt) {
            int col = nc * 32 + tt * 16 + lrow;
            float bias = b3[col];
            #pragma unroll
            for (int i = 0; i < 8; ++i) {
                float vl = acc2[tt][i] + bias;
                out[(size_t)(r0 + mbase + i) * OUTC + col] = fmaxf(vl, 0.f);
            }
        }
    }
}

// ---------------- launch ----------------
extern "C" void kernel_launch(void* const* d_in, const int* in_sizes, int n_in,
                              void* d_out, int out_size, void* d_ws, size_t ws_size,
                              hipStream_t stream) {
    (void)in_sizes; (void)n_in; (void)out_size; (void)ws_size;
    const float* X0    = (const float*)d_in[0];
    const float* Xs    = (const float*)d_in[1];
    const float* Wq    = (const float*)d_in[2];
    const float* bq    = (const float*)d_in[3];
    const float* Wk    = (const float*)d_in[4];
    // d_in[5] = bk: cancels inside softmax, unused
    const float* Wv    = (const float*)d_in[6];
    const float* bv    = (const float*)d_in[7];
    const float* gamma = (const float*)d_in[8];
    const float* W1    = (const float*)d_in[9];
    const float* b1    = (const float*)d_in[10];
    const float* W2    = (const float*)d_in[11];
    const float* b2    = (const float*)d_in[12];
    const float* W3    = (const float*)d_in[13];
    const float* b3    = (const float*)d_in[14];

    // workspace layout (16B-aligned offsets)
    char* ws = (char*)d_ws;
    float*    zbuf  = (float*)(ws + 0);              // 1024 f32
    float*    XV    = (float*)(ws + 4096);           // [512,512] f32: rows 0-255=X0, 256-511=V0
    float*    UW    = (float*)(ws + 1052672);        // [512,1024] f32: rows 0-255=u, 256-511=w
    float*    q0all = (float*)(ws + 3149824);        // [256,64]
    float*    k0all = (float*)(ws + 3215360);        // [256,64]
    float*    rowA  = (float*)(ws + 3280896);        // [65536]
    float*    rowAl = (float*)(ws + 3543040);        // [65536]
    _Float16* W2h   = (_Float16*)(ws + 3805184);     // [1024,512] f16
    _Float16* W3h   = (_Float16*)(ws + 4853760);     // [512,128] f16

    // weight conversions
    f32_to_f16_kernel<<<(H1 * H2 + 255) / 256, 256, 0, stream>>>(W2, W2h, H1 * H2);
    f32_to_f16_kernel<<<(H2 * OUTC + 255) / 256, 256, 0, stream>>>(W3, W3h, H2 * OUTC);

    // stack [X0; V0] then one GEMM against W1 -> u,w (weights read O(1) times)
    copy_f32_kernel<<<(BB * DD + 255) / 256, 256, 0, stream>>>(X0, XV, BB * DD);
    gemm_f32_kernel<<<(BB / 64) * (DD / 64), 256, 0, stream>>>(X0, Wv, XV + BB * DD, BB, DD, DD);
    gemm_f32_kernel<<<(BB / 64) * (DQK / 64), 256, 0, stream>>>(X0, Wq, q0all, BB, DQK, DD);
    gemm_f32_kernel<<<(BB / 64) * (DQK / 64), 256, 0, stream>>>(X0, Wk, k0all, BB, DQK, DD);
    gemm_f32_kernel<<<(2 * BB / 64) * (H1 / 64), 256, 0, stream>>>(XV, W1, UW, 2 * BB, H1, DD);

    z_kernel<<<H1 / 256, 256, 0, stream>>>(bv, W1, b1, gamma, zbuf);
    attn_scalar_kernel<<<BB, 256, 0, stream>>>(Xs, q0all, k0all, bq, gamma, rowA, rowAl);

    fused_mlp_kernel<<<(BB * NN) / 32, 256, 0, stream>>>(UW, zbuf, rowA, rowAl,
                                                         W2h, b2, W3h, b3, (float*)d_out);
}
